// DGN_19215683682387
// MI455X (gfx1250) — matrix operand order repaired and checked by reading, not compiled, not run on hardware
//
#include <hip/hip_runtime.h>
#include <math.h>

// ---------------- problem constants (match reference) ----------------
#define NN 20000
#define EE 320000
#define BB 64
#define NODE_DIM 64
#define EDGE_DIM 32
#define HD 128
#define TT 256
#define LL 5
#define DELTA 1.0f
#define PRE_K (2 * HD + EDGE_DIM)    // 288
#define POST_K (HD + 12 * HD)        // 1664
#define NTILES (HD / 16)             // 8
#define PRE_KT (PRE_K / 32)          // 9
#define POST_KT (POST_K / 32)        // 52
#define IN_KT (NODE_DIM / 32)        // 2
// swizzled-weight element counts (per layer)
#define SW_PER_KT (NTILES * 32 * 16) // 4096
#define PRE_SW (PRE_KT * SW_PER_KT)  // 36864
#define POST_SW (POST_KT * SW_PER_KT)// 212992
#define IN_SW (IN_KT * SW_PER_KT)    // 8192

typedef _Float16 v16h __attribute__((ext_vector_type(16)));
typedef _Float16 v8h  __attribute__((ext_vector_type(8)));
typedef float    v8f  __attribute__((ext_vector_type(8)));

// ---------------- WMMA helpers (CDNA5 16x16x32 f16 -> f32) ----------------
__device__ __forceinline__ v8f wmma_f16(v16h a, v16h b, v8f c) {
    return __builtin_amdgcn_wmma_f32_16x16x32_f16(false, a, false, b, (short)0, c,
                                                  false, false);
}

// A-fragment from LDS (row-major f16, lda halves/row, lda*2 % 16 == 0).
// K pattern = two contiguous 8-half chunks at +0/+16 (lane>=16 adds +8).
__device__ __forceinline__ v16h a_frag_lds(const _Float16* As, int lda, int kbase,
                                           int lane) {
    const int m  = lane & 15;
    const int hk = (lane >= 16) ? 8 : 0;
    const _Float16* p = As + m * lda + kbase + hk;
    v8h lo = *(const v8h*)p;
    v8h hi = *(const v8h*)(p + 16);
    return __builtin_shufflevector(lo, hi, 0, 1, 2, 3, 4, 5, 6, 7,
                                   8, 9, 10, 11, 12, 13, 14, 15);
}

// B-fragment from pre-swizzled f16 weights: one 32-byte contiguous load per lane.
__device__ __forceinline__ v16h b_frag_sw(const _Float16* __restrict__ Bsw,
                                          int ktile, int ntile, int lane) {
    return *(const v16h*)(Bsw + ((((size_t)ktile * NTILES + ntile) * 32 + lane) << 4));
}

// ---------------- weight swizzle: f32 [K,128] -> f16 B-fragment order ----------------
// out[((ktile*8 + ntile)*32 + lane)*16 + e] = W[(ktile*32 + (e&7) + 8*(lane>=16) + 16*(e>=8))*128
//                                               + ntile*16 + lane%16]
__global__ void dgn_swizzle_w(const float* __restrict__ W, _Float16* __restrict__ out,
                              int ktiles) {
    int o = blockIdx.x * blockDim.x + threadIdx.x;
    if (o >= ktiles * SW_PER_KT) return;
    int e     = o & 15;
    int lane  = (o >> 4) & 31;
    int ntile = (o >> 9) & 7;
    int ktile = o >> 12;
    int n = ntile * 16 + (lane & 15);
    int k = ktile * 32 + (e & 7) + ((lane >= 16) ? 8 : 0) + ((e >= 8) ? 16 : 0);
    out[o] = (_Float16)W[k * HD + n];
}

// ---------------- small utility kernels ----------------
__global__ void dgn_fill_u32(unsigned* __restrict__ p, unsigned val, int n) {
    int i = blockIdx.x * blockDim.x + threadIdx.x;
    if (i < n) p[i] = val;
}

__global__ void dgn_degree(const int* __restrict__ dst, float* __restrict__ deg) {
    int e = blockIdx.x * blockDim.x + threadIdx.x;
    if (e < EE) atomicAdd(&deg[dst[e]], 1.0f);
}

__global__ void dgn_scalers(const float* __restrict__ deg, float* __restrict__ inv_deg,
                            float* __restrict__ amp, float* __restrict__ att) {
    int n = blockIdx.x * blockDim.x + threadIdx.x;
    if (n >= NN) return;
    float d = deg[n];
    inv_deg[n] = (d > 0.0f) ? (1.0f / fmaxf(d, 1.0f)) : 0.0f;
    float logd = logf(d + 1.0f);
    amp[n] = logd / DELTA;
    att[n] = (logd > 0.0f) ? (DELTA / fmaxf(logd, 1e-12f)) : 1.0f;
}

// ---------------- input MLP: h = relu(node_feat @ W_in + b_in) ----------------
__global__ __launch_bounds__(256) void dgn_input_mlp(
    const float* __restrict__ nf, const _Float16* __restrict__ Bsw,
    const float* __restrict__ bias, float* __restrict__ h) {
    __shared__ _Float16 As[16 * NODE_DIM];
    const int nb = blockIdx.x * 16;
    const int tid = threadIdx.x;
    for (int i = tid; i < 16 * NODE_DIM; i += 256) {
        int r = i >> 6, c = i & 63;
        As[i] = (_Float16)nf[(nb + r) * NODE_DIM + c];
    }
    __syncthreads();
    const int wave = tid >> 5, lane = tid & 31;
    const int n0 = wave * 16;
    v8f acc = {};
#pragma unroll
    for (int kt = 0; kt < IN_KT; ++kt) {
        v16h b = b_frag_sw(Bsw, kt, wave, lane);
        v16h a = a_frag_lds(As, NODE_DIM, kt * 32, lane);
        acc = wmma_f16(a, b, acc);
    }
    const int mo = (lane >= 16) ? 8 : 0;
    const int n  = n0 + (lane & 15);
    const float bv = bias[n];
#pragma unroll
    for (int v = 0; v < 8; ++v) {
        h[(nb + v + mo) * HD + n] = fmaxf(acc[v] + bv, 0.0f);
    }
}

// ---------------- pretrans + fused segment scatter ----------------
// Block = 64 edges x 128 cols; 8 waves; wave w owns N-tile w and 4 M-tiles,
// reusing one B fragment across the 4 WMMAs per K-step.
#define PT_ROWS 64
__global__ __launch_bounds__(256) void dgn_pretrans_scatter(
    const float* __restrict__ h, const float* __restrict__ ef,
    const _Float16* __restrict__ Bsw, const float* __restrict__ bias,
    const int* __restrict__ src, const int* __restrict__ dst,
    float* __restrict__ s_buf, int* __restrict__ mx_buf, int* __restrict__ mn_buf) {
    __shared__ _Float16 As[PT_ROWS * PRE_K];   // 36 KB
    __shared__ int dst_s[PT_ROWS];
    const int eb = blockIdx.x * PT_ROWS;
    const int tid = threadIdx.x;
    if (tid < PT_ROWS) dst_s[tid] = dst[eb + tid];
    for (int i = tid; i < PT_ROWS * PRE_K; i += 256) {
        int r = i / PRE_K, c = i - r * PRE_K;
        int e = eb + r;
        float v;
        if (c < HD)          v = h[src[e] * HD + c];
        else if (c < 2 * HD) v = h[dst[e] * HD + (c - HD)];
        else                 v = ef[e * EDGE_DIM + (c - 2 * HD)];
        As[i] = (_Float16)v;
    }
    __syncthreads();
    const int wave = tid >> 5, lane = tid & 31;
    v8f acc[4] = {};
    for (int kt = 0; kt < PRE_KT; ++kt) {
        v16h b = b_frag_sw(Bsw, kt, wave, lane);
#pragma unroll
        for (int t = 0; t < 4; ++t) {
            v16h a = a_frag_lds(As + t * 16 * PRE_K, PRE_K, kt * 32, lane);
            acc[t] = wmma_f16(a, b, acc[t]);
        }
    }
    const int mo = (lane >= 16) ? 8 : 0;
    const int n  = wave * 16 + (lane & 15);
    const float bv = bias[n];
#pragma unroll
    for (int t = 0; t < 4; ++t) {
#pragma unroll
        for (int v = 0; v < 8; ++v) {
            float val = fmaxf(acc[t][v] + bv, 0.0f);     // m >= 0
            int d = dst_s[t * 16 + v + mo];
            int bits = __float_as_int(val);              // monotone for val >= 0
            atomicAdd(&s_buf[d * HD + n], val);
            atomicMax(&mx_buf[d * HD + n], bits);
            atomicMin(&mn_buf[d * HD + n], bits);
        }
    }
}

// ---------------- posttrans: h += relu(X @ qW + qb), X built on the fly ----------------
// Block = 32 nodes x 128 cols via dynamic LDS (32*1664*2 = 104 KB of 320 KB/WGP);
// wave w owns N-tile w and 2 M-tiles sharing each B fragment.
#define QT_ROWS 32
__global__ __launch_bounds__(256) void dgn_posttrans(
    float* __restrict__ h, const float* __restrict__ s_buf,
    const int* __restrict__ mx_buf, const int* __restrict__ mn_buf,
    const float* __restrict__ deg, const float* __restrict__ inv_deg,
    const float* __restrict__ amp, const float* __restrict__ att,
    const _Float16* __restrict__ Bsw, const float* __restrict__ bias) {
    extern __shared__ _Float16 Xs[];
    const int nb = blockIdx.x * QT_ROWS;
    const int tid = threadIdx.x;
    for (int i = tid; i < QT_ROWS * POST_K; i += 256) {
        int r = i / POST_K, c = i - r * POST_K;
        int node = nb + r;
        float v;
        if (c < HD) {
            v = h[node * HD + c];
        } else {
            int cc = c - HD;
            int g = cc >> 9;          // 0: *1, 1: *amp, 2: *att
            int w = cc & 511;
            int a = w >> 7;           // 0:mean 1:max 2:min 3:sum
            int col = w & 127;
            float sv = s_buf[node * HD + col];
            float x;
            if (a == 0)      x = sv * inv_deg[node];
            else if (a == 1) x = __int_as_float(mx_buf[node * HD + col]); // init 0
            else if (a == 2) x = (deg[node] > 0.0f)
                                   ? __int_as_float(mn_buf[node * HD + col]) : 0.0f;
            else             x = sv;
            if (g == 1)      x *= amp[node];
            else if (g == 2) x *= att[node];
            v = x;
        }
        Xs[i] = (_Float16)v;
    }
    __syncthreads();
    const int wave = tid >> 5, lane = tid & 31;
    v8f acc[2] = {};
    for (int kt = 0; kt < POST_KT; ++kt) {
        v16h b = b_frag_sw(Bsw, kt, wave, lane);
#pragma unroll
        for (int t = 0; t < 2; ++t) {
            v16h a = a_frag_lds(Xs + t * 16 * POST_K, POST_K, kt * 32, lane);
            acc[t] = wmma_f16(a, b, acc[t]);
        }
    }
    const int mo = (lane >= 16) ? 8 : 0;
    const int n  = wave * 16 + (lane & 15);
    const float bv = bias[n];
#pragma unroll
    for (int t = 0; t < 2; ++t) {
#pragma unroll
        for (int v = 0; v < 8; ++v) {
            int row = nb + t * 16 + v + mo;
            h[row * HD + n] += fmaxf(acc[t][v] + bv, 0.0f);   // residual
        }
    }
}

// ---------------- readout ----------------
__global__ void dgn_readout_scatter(const float* __restrict__ h,
                                    const int* __restrict__ n2g,
                                    float* __restrict__ g_sum, int* __restrict__ g_max,
                                    float* __restrict__ g_cnt) {
    int idx = blockIdx.x * blockDim.x + threadIdx.x;
    if (idx >= NN * HD) return;
    int node = idx >> 7, c = idx & 127;
    int g = n2g[node];
    float v = h[idx];
    atomicAdd(&g_sum[g * HD + c], v);
    atomicMax(&g_max[g * HD + c], __float_as_int(v));  // h >= 0
    if (c == 0) atomicAdd(&g_cnt[g], 1.0f);
}

__global__ void dgn_out1(const float* __restrict__ g_sum, const int* __restrict__ g_max,
                         const float* __restrict__ g_cnt, const float* __restrict__ W1,
                         const float* __restrict__ b1, float* __restrict__ r1) {
    int idx = blockIdx.x * blockDim.x + threadIdx.x;   // BB*HD
    if (idx >= BB * HD) return;
    int b = idx >> 7, j = idx & 127;
    float inv = 1.0f / fmaxf(g_cnt[b], 1.0f);
    float acc = b1[j];
    for (int k = 0; k < HD; ++k) {
        float sum = g_sum[b * HD + k];
        float mx  = __int_as_float(g_max[b * HD + k]);
        acc += (sum * inv) * W1[k * HD + j]
             + mx          * W1[(HD + k) * HD + j]
             + sum         * W1[(2 * HD + k) * HD + j];
    }
    r1[idx] = fmaxf(acc, 0.0f);
}

__global__ void dgn_out2(const float* __restrict__ r1, const float* __restrict__ W2,
                         const float* __restrict__ b2, float* __restrict__ out) {
    int idx = blockIdx.x * blockDim.x + threadIdx.x;   // BB*TT
    if (idx >= BB * TT) return;
    int b = idx >> 8, t = idx & 255;
    float acc = b2[t];
    for (int k = 0; k < HD; ++k) acc += r1[b * HD + k] * W2[k * TT + t];
    out[idx] = acc;
}

// ---------------- host launcher ----------------
extern "C" void kernel_launch(void* const* d_in, const int* in_sizes, int n_in,
                              void* d_out, int out_size, void* d_ws, size_t ws_size,
                              hipStream_t stream) {
    (void)in_sizes; (void)n_in; (void)out_size; (void)ws_size;
    const float* node_feat = (const float*)d_in[0];
    const float* edge_feat = (const float*)d_in[1];
    const float* W_in      = (const float*)d_in[2];
    const float* b_in      = (const float*)d_in[3];
    const float* pre_W     = (const float*)d_in[4];
    const float* pre_b     = (const float*)d_in[5];
    const float* post_W    = (const float*)d_in[6];
    const float* post_b    = (const float*)d_in[7];
    const float* out_W1    = (const float*)d_in[8];
    const float* out_b1    = (const float*)d_in[9];
    const float* out_W2    = (const float*)d_in[10];
    const float* out_b2    = (const float*)d_in[11];
    const int*   src       = (const int*)d_in[12];
    const int*   dst       = (const int*)d_in[13];
    const int*   n2g       = (const int*)d_in[14];
    float* out = (float*)d_out;

    // workspace carve
    char* cur = (char*)d_ws;
    auto alloc = [&](size_t bytes) {
        void* p = (void*)cur;
        cur += (bytes + 255) & ~(size_t)255;
        return p;
    };
    float*     h       = (float*)alloc((size_t)NN * HD * 4);
    float*     s_buf   = (float*)alloc((size_t)NN * HD * 4);
    int*       mx_buf  = (int*)  alloc((size_t)NN * HD * 4);
    int*       mn_buf  = (int*)  alloc((size_t)NN * HD * 4);
    float*     deg     = (float*)alloc((size_t)NN * 4);
    float*     inv_deg = (float*)alloc((size_t)NN * 4);
    float*     amp     = (float*)alloc((size_t)NN * 4);
    float*     att     = (float*)alloc((size_t)NN * 4);
    float*     g_sum   = (float*)alloc((size_t)BB * HD * 4);
    int*       g_max   = (int*)  alloc((size_t)BB * HD * 4);
    float*     g_cnt   = (float*)alloc((size_t)BB * 4);
    float*     r1      = (float*)alloc((size_t)BB * HD * 4);
    _Float16*  win_h   = (_Float16*)alloc((size_t)IN_SW * 2);
    _Float16*  wpre_h  = (_Float16*)alloc((size_t)LL * PRE_SW * 2);
    _Float16*  wpost_h = (_Float16*)alloc((size_t)LL * POST_SW * 2);

    const unsigned POS_INF = 0x7F800000u;
    const int TB = 256;
    auto blocks = [](int n, int tb) { return (n + tb - 1) / tb; };

    // --- weight pre-swizzle to f16 WMMA fragment order ---
    dgn_swizzle_w<<<blocks(IN_SW, TB), TB, 0, stream>>>(W_in, win_h, IN_KT);
    for (int l = 0; l < LL; ++l) {
        dgn_swizzle_w<<<blocks(PRE_SW, TB), TB, 0, stream>>>(
            pre_W + (size_t)l * PRE_K * HD, wpre_h + (size_t)l * PRE_SW, PRE_KT);
        dgn_swizzle_w<<<blocks(POST_SW, TB), TB, 0, stream>>>(
            post_W + (size_t)l * POST_K * HD, wpost_h + (size_t)l * POST_SW, POST_KT);
    }

    // --- degrees + scalers ---
    dgn_fill_u32<<<blocks(NN, TB), TB, 0, stream>>>((unsigned*)deg, 0u, NN);
    dgn_degree<<<blocks(EE, TB), TB, 0, stream>>>(dst, deg);
    dgn_scalers<<<blocks(NN, TB), TB, 0, stream>>>(deg, inv_deg, amp, att);

    // --- input MLP ---
    dgn_input_mlp<<<NN / 16, TB, 0, stream>>>(node_feat, win_h, b_in, h);

    // --- message-passing layers ---
    for (int l = 0; l < LL; ++l) {
        dgn_fill_u32<<<blocks(NN * HD, TB), TB, 0, stream>>>((unsigned*)s_buf, 0u, NN * HD);
        dgn_fill_u32<<<blocks(NN * HD, TB), TB, 0, stream>>>((unsigned*)mx_buf, 0u, NN * HD);
        dgn_fill_u32<<<blocks(NN * HD, TB), TB, 0, stream>>>((unsigned*)mn_buf, POS_INF, NN * HD);
        dgn_pretrans_scatter<<<EE / PT_ROWS, TB, 0, stream>>>(
            h, edge_feat, wpre_h + (size_t)l * PRE_SW, pre_b + (size_t)l * HD,
            src, dst, s_buf, mx_buf, mn_buf);
        dgn_posttrans<<<NN / QT_ROWS, TB, (size_t)QT_ROWS * POST_K * 2, stream>>>(
            h, s_buf, mx_buf, mn_buf, deg, inv_deg, amp, att,
            wpost_h + (size_t)l * POST_SW, post_b + (size_t)l * HD);
    }

    // --- readout ---
    dgn_fill_u32<<<blocks(BB * HD, TB), TB, 0, stream>>>((unsigned*)g_sum, 0u, BB * HD);
    dgn_fill_u32<<<blocks(BB * HD, TB), TB, 0, stream>>>((unsigned*)g_max, 0u, BB * HD);
    dgn_fill_u32<<<blocks(BB, TB), TB, 0, stream>>>((unsigned*)g_cnt, 0u, BB);
    dgn_readout_scatter<<<blocks(NN * HD, TB), TB, 0, stream>>>(h, n2g, g_sum, g_max, g_cnt);
    dgn_out1<<<blocks(BB * HD, TB), TB, 0, stream>>>(g_sum, g_max, g_cnt, out_W1, out_b1, r1);
    dgn_out2<<<blocks(BB * TT, TB), TB, 0, stream>>>(r1, out_W2, out_b2, out);
}
